// CrossModalPatchXAttnBlock_29892972380242
// MI455X (gfx1250) — matrix operand order repaired and checked by reading, not compiled, not run on hardware
//
#include <hip/hip_runtime.h>
#include <math.h>

#define DEV __device__ __forceinline__

typedef __attribute__((ext_vector_type(16))) __bf16 v16bf;
typedef __attribute__((ext_vector_type(8)))  float  v8f;
typedef unsigned short u16;
typedef unsigned int   u32;

union AF { v16bf v; u32 u[8]; };   // 16 bf16 = 8 VGPRs (A or B fragment)
union CF { v8f  v; float f[8]; };  // 16x16 f32 C/D accumulator

DEV u16 f2bf(float x) {            // f32 -> bf16 round-to-nearest-even
  u32 u = __float_as_uint(x);
  u32 r = u + 0x7FFFu + ((u >> 16) & 1u);
  return (u16)(r >> 16);
}
DEV u32 pack2(float a, float b) {
  return (u32)f2bf(a) | ((u32)f2bf(b) << 16);
}
DEV float gelu_f(float x) { return 0.5f * x * (1.0f + erff(x * 0.70710678118654752f)); }

// ---------------------------------------------------------------------------
// Async 16B global->LDS copy (CDNA5 GLOBAL_LOAD_ASYNC_TO_LDS_B128, ASYNCcnt)
// with a synchronous b128 fallback if the builtin is unavailable.
// Builtin signature (from compiler diagnostic): (v4i32 AS1*, v4i32 AS3*, i32, i32).
// ---------------------------------------------------------------------------
#if defined(__has_builtin)
#if __has_builtin(__builtin_amdgcn_global_load_async_to_lds_b128)
#define ASYNC_LDS 1
#endif
#endif

typedef int v4i __attribute__((__vector_size__(16)));
typedef __attribute__((address_space(1))) v4i as1_v4i;
typedef __attribute__((address_space(3))) v4i as3_v4i;

DEV void cp16(u16* l, const u16* g) {
#ifdef ASYNC_LDS
  __builtin_amdgcn_global_load_async_to_lds_b128((as1_v4i*)g, (as3_v4i*)l, 0, 0);
#else
  *(uint4*)l = *(const uint4*)g;
#endif
}
DEV void async_drain() {
#ifdef ASYNC_LDS
#if __has_builtin(__builtin_amdgcn_s_wait_asynccnt)
  __builtin_amdgcn_s_wait_asynccnt(0);
#else
  asm volatile("s_wait_asynccnt 0" ::: "memory");
#endif
#endif
}

// ---------------------------------------------------------------------------
// One-time weight prep: W[K,N] f32  ->  Wt[N,K] bf16 (n-major), tiled 32x32
// transpose through LDS so both global phases stay coalesced.
// ---------------------------------------------------------------------------
__launch_bounds__(256)
__global__ void wtrans_kernel(const float* __restrict__ W, u16* __restrict__ Wt,
                              int K, int N) {
  __shared__ u16 T[32][33];
  const int n0 = blockIdx.x * 32, k0 = blockIdx.y * 32;
  const int t = threadIdx.x;
  const int c = t & 31, r0 = t >> 5;
  #pragma unroll
  for (int i = 0; i < 4; ++i) {
    int r = r0 + i * 8;
    T[c][r] = f2bf(W[(size_t)(k0 + r) * N + n0 + c]);
  }
  __syncthreads();
  #pragma unroll
  for (int p = 0; p < 2; ++p) {
    int idx = p * 256 + t;
    int rn = idx >> 4, ck = idx & 15;
    u32 v = (u32)T[rn][2 * ck] | ((u32)T[rn][2 * ck + 1] << 16);
    *(u32*)(Wt + (size_t)(n0 + rn) * K + k0 + 2 * ck) = v;
  }
}

// ---------------------------------------------------------------------------
// LayerNorm over D=768 (f32 in, bf16 out), one row per block, 256 threads.
// ---------------------------------------------------------------------------
__launch_bounds__(256)
__global__ void ln_kernel(const float* __restrict__ x, const float* __restrict__ g,
                          const float* __restrict__ b, u16* __restrict__ y) {
  const int row = blockIdx.x;
  const float* xr = x + (size_t)row * 768;
  u16* yr = y + (size_t)row * 768;
  const int tid = threadIdx.x;
  const int wv = tid >> 5, ln = tid & 31;

  float v0 = xr[tid], v1 = xr[tid + 256], v2 = xr[tid + 512];
  float s = v0 + v1 + v2;
  __shared__ float sb[8];
  __shared__ float mean_s, rstd_s;
  #pragma unroll
  for (int o = 16; o; o >>= 1) s += __shfl_xor(s, o, 32);
  if (!ln) sb[wv] = s;
  __syncthreads();
  if (tid == 0) {
    float t = 0.f;
    #pragma unroll
    for (int i = 0; i < 8; ++i) t += sb[i];
    mean_s = t * (1.0f / 768.0f);
  }
  __syncthreads();
  const float mu = mean_s;
  float d0 = v0 - mu, d1 = v1 - mu, d2 = v2 - mu;
  float q = d0 * d0 + d1 * d1 + d2 * d2;
  #pragma unroll
  for (int o = 16; o; o >>= 1) q += __shfl_xor(q, o, 32);
  if (!ln) sb[wv] = q;
  __syncthreads();
  if (tid == 0) {
    float t = 0.f;
    #pragma unroll
    for (int i = 0; i < 8; ++i) t += sb[i];
    rstd_s = rsqrtf(t * (1.0f / 768.0f) + 1e-5f);
  }
  __syncthreads();
  const float rs = rstd_s;
  yr[tid]       = f2bf(d0 * rs * g[tid]       + b[tid]);
  yr[tid + 256] = f2bf(d1 * rs * g[tid + 256] + b[tid + 256]);
  yr[tid + 512] = f2bf(d2 * rs * g[tid + 512] + b[tid + 512]);
}

// ---------------------------------------------------------------------------
// Fragment gathers from LDS per CDNA5 WMMA VGPR layouts (ISA 7.12.2).
// ---------------------------------------------------------------------------
DEV void load_afrag(const u16* buf, int row0, int stride, AF& f) {
  const int lane = threadIdx.x & 31;
  const int m = lane & 15, half = lane >> 4;
  const u16* p = buf + (row0 + m) * stride;
  #pragma unroll
  for (int j = 0; j < 8; ++j) {
    int k = (j < 4 ? 2 * j : 16 + 2 * (j - 4)) + half * 8;
    f.u[j] = *(const u32*)&p[k];
  }
}
DEV void load_bfrag(const u16* buf, int col0, int stride, AF& f) {
  const int lane = threadIdx.x & 31;
  const int n = col0 + (lane & 15), half = lane >> 4;
  const u16* p = buf + n * stride;
  #pragma unroll
  for (int j = 0; j < 8; ++j) {
    int k = (j < 4 ? 2 * j : 16 + 2 * (j - 4)) + half * 8;
    f.u[j] = *(const u32*)&p[k];
  }
}

// ---------------------------------------------------------------------------
// GEMM: C[4096 x Nout] = act(A_bf16[4096 x K] @ Wt_bf16[Nout x K]^T + bias)
// Block 128x128, BK=32, 8 waves, wave tile 64x32 (4x2 WMMA accums).
// Double-buffered LDS; staging = async b128 LDS fills (drained with
// s_wait_asynccnt before the publishing barrier).
// outmode: 0 = f32 out (+optional residual), 1 = bf16 out, 2 = bf16 transposed
// out [Nout x 4096] (used for V so flash can stage V with b128 copies).
// ---------------------------------------------------------------------------
DEV void gemm_stage(const u16* __restrict__ A, const u16* __restrict__ Wt, int K,
                    int m0, int n0, int kt, u16* As, u16* Bs, int tid) {
  #pragma unroll
  for (int p = 0; p < 2; ++p) {
    int idx = p * 256 + tid;
    int row = idx >> 2, ch = idx & 3;
    cp16(As + row * 40 + ch * 8, A  + (size_t)(m0 + row) * K + kt + ch * 8);
    cp16(Bs + row * 40 + ch * 8, Wt + (size_t)(n0 + row) * K + kt + ch * 8);
  }
}

__launch_bounds__(256)
__global__ void gemm_bf16(const u16* __restrict__ A, const u16* __restrict__ Wt,
                          const float* __restrict__ bias, const float* __restrict__ res,
                          void* __restrict__ Cout, int K, int Nout, int act, int outmode) {
  __shared__ __align__(16) u16 As[2][128 * 40];
  __shared__ __align__(16) u16 Bs[2][128 * 40];
  const int tid = threadIdx.x;
  const int lane = tid & 31, wave = tid >> 5;
  const int wm = (wave >> 2) * 64, wn = (wave & 3) * 32;
  const int m0 = blockIdx.y * 128, n0 = blockIdx.x * 128;

  CF acc[4][2];
  #pragma unroll
  for (int mt = 0; mt < 4; ++mt)
    #pragma unroll
    for (int nt = 0; nt < 2; ++nt)
      #pragma unroll
      for (int e = 0; e < 8; ++e) acc[mt][nt].f[e] = 0.0f;

  gemm_stage(A, Wt, K, m0, n0, 0, As[0], Bs[0], tid);
  int cur = 0;
  for (int kt = 0; kt < K; kt += 32) {
    async_drain();
    __syncthreads();
    if (kt + 32 < K)
      gemm_stage(A, Wt, K, m0, n0, kt + 32, As[cur ^ 1], Bs[cur ^ 1], tid);

    AF af[4], bf2[2];
    #pragma unroll
    for (int mt = 0; mt < 4; ++mt) load_afrag(As[cur], wm + mt * 16, 40, af[mt]);
    #pragma unroll
    for (int nt = 0; nt < 2; ++nt) load_bfrag(Bs[cur], wn + nt * 16, 40, bf2[nt]);
    #pragma unroll
    for (int mt = 0; mt < 4; ++mt)
      #pragma unroll
      for (int nt = 0; nt < 2; ++nt)
        acc[mt][nt].v = __builtin_amdgcn_wmma_f32_16x16x32_bf16(
            false, af[mt].v, false, bf2[nt].v, (short)0, acc[mt][nt].v, false, false);
    cur ^= 1;
  }

  const int half = lane >> 4, nn = lane & 15;
  #pragma unroll
  for (int mt = 0; mt < 4; ++mt)
    #pragma unroll
    for (int nt = 0; nt < 2; ++nt) {
      int col = n0 + wn + nt * 16 + nn;
      float bc = bias[col];
      float vv[8];
      #pragma unroll
      for (int r = 0; r < 8; ++r) {
        float v = acc[mt][nt].f[r] + bc;
        if (act) v = gelu_f(v);
        vv[r] = v;
      }
      if (outmode == 0) {
        float* C = (float*)Cout;
        #pragma unroll
        for (int r = 0; r < 8; ++r) {
          int row = m0 + wm + mt * 16 + r + half * 8;
          float v = vv[r];
          if (res) v += res[(size_t)row * Nout + col];
          C[(size_t)row * Nout + col] = v;
        }
      } else if (outmode == 1) {
        u16* C = (u16*)Cout;
        #pragma unroll
        for (int r = 0; r < 8; ++r) {
          int row = m0 + wm + mt * 16 + r + half * 8;
          C[(size_t)row * Nout + col] = f2bf(vv[r]);
        }
      } else {  // transposed bf16: [Nout x 4096], rows r..r+7 contiguous
        u16* Ct = (u16*)Cout;
        int rowb = m0 + wm + mt * 16 + half * 8;
        uint4 q = make_uint4(pack2(vv[0], vv[1]), pack2(vv[2], vv[3]),
                             pack2(vv[4], vv[5]), pack2(vv[6], vv[7]));
        *(uint4*)(Ct + (size_t)col * 4096 + rowb) = q;
      }
    }
}

// ---------------------------------------------------------------------------
// Flash attention (wave32, WMMA). blockIdx.x = (b,h), blockIdx.y = 64-q-row
// slab; 4 waves x 16 q-rows. Q/K bf16 row-major, V pre-transposed bf16
// [hd, token] so all staging is async b128 LDS fills. Double-buffered K/V.
// Online softmax; P re-laid out via wave-private LDS bounce. bf16 output.
// ---------------------------------------------------------------------------
DEV void kv_stage(const u16* __restrict__ kp, const u16* __restrict__ vt,
                  int bb, int h, int kvs, int key0, u16* Ks, u16* Vs, int tid) {
  #pragma unroll
  for (int p = 0; p < 2; ++p) {
    int idx = p * 128 + tid;
    int krow = idx >> 3, kch = idx & 7;
    cp16(Ks + krow * 72 + kch * 8,
         kp + (size_t)(bb * 1024 + key0 + krow) * kvs + h * 64 + kch * 8);
    int hd = idx >> 2, vch = idx & 3;
    cp16(Vs + hd * 40 + vch * 8,
         vt + (size_t)(h * 64 + hd) * 4096 + bb * 1024 + key0 + vch * 8);
  }
}

__launch_bounds__(128)
__global__ void flash_attn_wmma(const u16* __restrict__ qp, const u16* __restrict__ kp,
                                const u16* __restrict__ vt, u16* __restrict__ outp,
                                int qs, int kvs, float scale) {
  __shared__ __align__(16) u16 Qs[64 * 72];
  __shared__ __align__(16) u16 Ks[2][32 * 72];
  __shared__ __align__(16) u16 Vs[2][64 * 40];     // [hd][key]
  __shared__ __align__(16) u16 Ps[4 * 16 * 40];    // per-wave P tile

  const int tid = threadIdx.x;
  const int lane = tid & 31, wave = tid >> 5;
  const int bb = blockIdx.x / 12, h = blockIdx.x % 12;
  const int q0 = blockIdx.y * 64;
  const int m = lane & 15, half = lane >> 4;

  // stage Q (64 x 64 bf16) once: 512 b128 chunks
  #pragma unroll
  for (int p = 0; p < 4; ++p) {
    int idx = p * 128 + tid;
    int row = idx >> 3, ch = idx & 7;
    cp16(Qs + row * 72 + ch * 8,
         qp + (size_t)(bb * 1024 + q0 + row) * qs + h * 64 + ch * 8);
  }
  kv_stage(kp, vt, bb, h, kvs, 0, Ks[0], Vs[0], tid);
  async_drain();
  __syncthreads();

  AF qf[2];
  #pragma unroll
  for (int c = 0; c < 2; ++c) {
    const u16* p = Qs + (wave * 16 + m) * 72;
    #pragma unroll
    for (int j = 0; j < 8; ++j) {
      int hd = c * 32 + (j < 4 ? 2 * j : 16 + 2 * (j - 4)) + half * 8;
      qf[c].u[j] = *(const u32*)&p[hd];
    }
  }

  float m_[8], l_[8], alph[8];
  CF o[4];
  #pragma unroll
  for (int r = 0; r < 8; ++r) { m_[r] = -3.0e38f; l_[r] = 0.0f; }
  #pragma unroll
  for (int c = 0; c < 4; ++c)
    #pragma unroll
    for (int e = 0; e < 8; ++e) o[c].f[e] = 0.0f;

  int cur = 0;
  for (int s = 0; s < 32; ++s) {
    if (s) { async_drain(); __syncthreads(); }
    if (s + 1 < 32) kv_stage(kp, vt, bb, h, kvs, (s + 1) * 32, Ks[cur ^ 1], Vs[cur ^ 1], tid);
    const u16* Kc = Ks[cur];
    const u16* Vc = Vs[cur];

    // S = scale * Q K^T for 32 keys (two 16-key accumulators)
    CF s0, s1;
    #pragma unroll
    for (int e = 0; e < 8; ++e) { s0.f[e] = 0.0f; s1.f[e] = 0.0f; }
    #pragma unroll
    for (int c = 0; c < 2; ++c) {
      AF kf0, kf1;
      #pragma unroll
      for (int j = 0; j < 8; ++j) {
        int hd = c * 32 + (j < 4 ? 2 * j : 16 + 2 * (j - 4)) + half * 8;
        kf0.u[j] = *(const u32*)&Kc[m * 72 + hd];
        kf1.u[j] = *(const u32*)&Kc[(16 + m) * 72 + hd];
      }
      s0.v = __builtin_amdgcn_wmma_f32_16x16x32_bf16(false, qf[c].v, false, kf0.v,
                                                     (short)0, s0.v, false, false);
      s1.v = __builtin_amdgcn_wmma_f32_16x16x32_bf16(false, qf[c].v, false, kf1.v,
                                                     (short)0, s1.v, false, false);
    }

    // online softmax over this 32-key tile
    #pragma unroll
    for (int r = 0; r < 8; ++r) {
      float a0 = s0.f[r] * scale, a1 = s1.f[r] * scale;
      float mx = fmaxf(a0, a1);
      #pragma unroll
      for (int off = 1; off < 16; off <<= 1) mx = fmaxf(mx, __shfl_xor(mx, off, 16));
      float mn = fmaxf(m_[r], mx);
      float al = __expf(m_[r] - mn);
      float p0 = __expf(a0 - mn), p1 = __expf(a1 - mn);
      float rs = p0 + p1;
      #pragma unroll
      for (int off = 1; off < 16; off <<= 1) rs += __shfl_xor(rs, off, 16);
      l_[r] = l_[r] * al + rs;
      m_[r] = mn;
      alph[r] = al;
      Ps[wave * 640 + (r + half * 8) * 40 + m]      = f2bf(p0);
      Ps[wave * 640 + (r + half * 8) * 40 + 16 + m] = f2bf(p1);
    }
    #pragma unroll
    for (int c = 0; c < 4; ++c)
      #pragma unroll
      for (int r = 0; r < 8; ++r) o[c].f[r] *= alph[r];

    asm volatile("" ::: "memory");  // keep wave-private DS write->read ordered

    AF pf;
    #pragma unroll
    for (int j = 0; j < 8; ++j) {
      int k = (j < 4 ? 2 * j : 16 + 2 * (j - 4)) + half * 8;
      pf.u[j] = *(const u32*)&Ps[wave * 640 + m * 40 + k];
    }
    #pragma unroll
    for (int c = 0; c < 4; ++c) {
      AF vf;
      #pragma unroll
      for (int j = 0; j < 8; ++j) {
        int key = (j < 4 ? 2 * j : 16 + 2 * (j - 4)) + half * 8;
        vf.u[j] = *(const u32*)&Vc[(c * 16 + m) * 40 + key];
      }
      o[c].v = __builtin_amdgcn_wmma_f32_16x16x32_bf16(false, pf.v, false, vf.v,
                                                       (short)0, o[c].v, false, false);
    }
    cur ^= 1;
  }

  #pragma unroll
  for (int c = 0; c < 4; ++c)
    #pragma unroll
    for (int r = 0; r < 8; ++r) {
      int row = q0 + wave * 16 + r + half * 8;
      float inv = 1.0f / l_[r];
      outp[(size_t)(bb * 1024 + row) * 768 + h * 64 + c * 16 + m] = f2bf(o[c].f[r] * inv);
    }
}

// ---------------------------------------------------------------------------
// Host orchestration
// ---------------------------------------------------------------------------
extern "C" void kernel_launch(void* const* d_in, const int* in_sizes, int n_in,
                              void* d_out, int out_size, void* d_ws, size_t ws_size,
                              hipStream_t stream) {
  (void)in_sizes; (void)n_in; (void)out_size; (void)ws_size;
  auto in = [&](int i) { return (const float*)d_in[i]; };
  char* base = (char*)d_ws;
  size_t off = 0;
  auto alloc = [&](size_t bytes) {
    void* p = base + off;
    off = (off + bytes + 255) & ~(size_t)255;
    return p;
  };
  const size_t SZ = 4096ull * 768ull;

  float* I1 = (float*)alloc(SZ * 4);
  float* E1 = (float*)alloc(SZ * 4);
  float* I2 = (float*)alloc(SZ * 4);
  float* E2 = (float*)alloc(SZ * 4);
  u16* HID = (u16*)alloc(4096ull * 3072 * 2);
  u16* QK  = (u16*)alloc(4096ull * 1536 * 2);   // self q|k fused, or cross q + k
  u16* VT  = (u16*)alloc(SZ * 2);               // transposed V [768][4096]
  u16* ATT = (u16*)alloc(SZ * 2);
  u16* LNA = (u16*)alloc(SZ * 2);
  u16* LNB = (u16*)alloc(SZ * 2);
  u16* LNC = (u16*)alloc(SZ * 2);
  u16* LND = (u16*)alloc(SZ * 2);

  // one-time weight transpose+convert: W[K,N] f32 -> Wt[N,K] bf16
  auto wt = [&](int idx, int K, int N) {
    u16* p = (u16*)alloc((size_t)K * N * 2);
    wtrans_kernel<<<dim3(N / 32, K / 32), dim3(256), 0, stream>>>(in(idx), p, K, N);
    return p;
  };
  u16* Wqkv_i  = wt(14, 768, 2304);
  u16* Wproj_i = wt(16, 768, 768);
  u16* Wqkv_e  = wt(18, 768, 2304);
  u16* Wproj_e = wt(20, 768, 768);
  u16* Wxei_q = wt(22, 768, 768), *Wxei_k = wt(24, 768, 768);
  u16* Wxei_v = wt(26, 768, 768), *Wxei_p = wt(28, 768, 768);
  u16* Wxie_q = wt(30, 768, 768), *Wxie_k = wt(32, 768, 768);
  u16* Wxie_v = wt(34, 768, 768), *Wxie_p = wt(36, 768, 768);
  u16* Wfc1_i = wt(38, 768, 3072), *Wfc2_i = wt(40, 3072, 768);
  u16* Wfc1_e = wt(42, 768, 3072), *Wfc2_e = wt(44, 3072, 768);

  auto ln = [&](const float* x, const float* g, const float* b, u16* y) {
    ln_kernel<<<dim3(4096), dim3(256), 0, stream>>>(x, g, b, y);
  };
  auto gemm = [&](const u16* A, const u16* W, const float* b, const float* r,
                  void* C, int K, int Nout, int act, int mode) {
    gemm_bf16<<<dim3(Nout / 128, 32), dim3(256), 0, stream>>>(A, W, b, r, C, K, Nout, act, mode);
  };
  auto attn = [&](const u16* q, const u16* k, const u16* v, u16* o,
                  int qs, int kvs, float sc) {
    flash_attn_wmma<<<dim3(48, 16), dim3(128), 0, stream>>>(q, k, v, o, qs, kvs, sc);
  };

  const float* img = in(0); const float* evt = in(1);
  float* OUTI = (float*)d_out;
  float* OUTE = OUTI + SZ;

  // self-attention (img): q|k fused GEMM + transposed-V GEMM
  ln(img, in(2), in(3), LNA);
  gemm(LNA, Wqkv_i,              in(15),        nullptr, QK, 768, 1536, 0, 1);
  gemm(LNA, Wqkv_i + 1536 * 768, in(15) + 1536, nullptr, VT, 768, 768, 0, 2);
  attn(QK, QK + 768, VT, ATT, 1536, 1536, 0.125f);
  gemm(ATT, Wproj_i, in(17), img, I1, 768, 768, 0, 0);
  // self-attention (evt)
  ln(evt, in(4), in(5), LNA);
  gemm(LNA, Wqkv_e,              in(19),        nullptr, QK, 768, 1536, 0, 1);
  gemm(LNA, Wqkv_e + 1536 * 768, in(19) + 1536, nullptr, VT, 768, 768, 0, 2);
  attn(QK, QK + 768, VT, ATT, 1536, 1536, 0.125f);
  gemm(ATT, Wproj_e, in(21), evt, E1, 768, 768, 0, 0);

  // cross LNs
  ln(I1, in(6), in(7), LNA);   // ln_q2(i1)
  ln(E1, in(8), in(9), LNB);   // ln_kv2(e1)
  ln(E1, in(6), in(7), LNC);   // ln_q2(e1)
  ln(I1, in(8), in(9), LND);   // ln_kv2(i1)

  // xei: img queries evt (negated scores)
  gemm(LNA, Wxei_q, in(23), nullptr, QK,      768, 768, 0, 1);
  gemm(LNB, Wxei_k, in(25), nullptr, QK + SZ, 768, 768, 0, 1);
  gemm(LNB, Wxei_v, in(27), nullptr, VT,      768, 768, 0, 2);
  attn(QK, QK + SZ, VT, ATT, 768, 768, -0.125f);
  gemm(ATT, Wxei_p, in(29), I1, I2, 768, 768, 0, 0);

  // xie: evt queries img (negated scores)
  gemm(LNC, Wxie_q, in(31), nullptr, QK,      768, 768, 0, 1);
  gemm(LND, Wxie_k, in(33), nullptr, QK + SZ, 768, 768, 0, 1);
  gemm(LND, Wxie_v, in(35), nullptr, VT,      768, 768, 0, 2);
  attn(QK, QK + SZ, VT, ATT, 768, 768, -0.125f);
  gemm(ATT, Wxie_p, in(37), E1, E2, 768, 768, 0, 0);

  // MLP (img)
  ln(I2, in(10), in(11), LNA);
  gemm(LNA, Wfc1_i, in(39), nullptr, HID, 768, 3072, 1, 1);
  gemm(HID, Wfc2_i, in(41), I2, OUTI, 3072, 768, 0, 0);
  // MLP (evt)
  ln(E2, in(12), in(13), LNA);
  gemm(LNA, Wfc1_e, in(43), nullptr, HID, 768, 3072, 1, 1);
  gemm(HID, Wfc2_e, in(45), E2, OUTE, 3072, 768, 0, 0);
}
